// InteractionBlock_43344809951436
// MI455X (gfx1250) — matrix operand order repaired and checked by reading, not compiled, not run on hardware
//
#include <hip/hip_runtime.h>

// ============================================================================
// InteractionBlock (ViT-Adapter injector/extractor) for MI455X / gfx1250.
//
// Precision: all GEMMs via v_wmma_f32_16x16x32_bf16 (bf16 in, fp32 accum).
// ~157 GFLOP of GEMM vs ~1.5 GB activation traffic at 23.3 TB/s -> near the
// balance point; bf16 WMMA is the rewarded pipe, fp32 accumulation keeps the
// LN/residual chain sane.  Weights (~24MB) + the value tensor (~66MB) fit in
// the 192MB L2, so the deformable-attention gathers are L2-resident.
//
// Data movement: GEMM tiles are staged fp32 into LDS with the CDNA5 async
// path (GLOBAL_LOAD_ASYNC_TO_LDS_B128, ASYNCcnt + s_wait_asynccnt), double
// buffered so the DMA of tile k+1 overlaps the WMMA work on tile k.  bf16
// conversion happens in-register (v_cvt) while building WMMA fragments.
// ============================================================================

#define DIM   768
#define BSZ   4
#define NT    1024          // tail queries (injector)
#define NX    1029
#define NC    5376          // context tokens (extractor queries / inj values)
#define FFN   1024
#define CCH   192           // conv-ffn hidden
#define NHEAD 6
#define HD    128

typedef __attribute__((ext_vector_type(16))) __bf16 v16bf;
typedef __attribute__((ext_vector_type(8)))  float  v8f;
typedef __attribute__((ext_vector_type(4)))  int    v4i_t;

__device__ __forceinline__ float gelu_f(float x) {
  return 0.5f * x * (1.0f + erff(x * 0.7071067811865476f));   // exact gelu
}

// ---- CDNA5 async global->LDS staging (guarded; falls back to sync copy) ----
#if defined(__gfx1250__) &&                                                    \
    __has_builtin(__builtin_amdgcn_global_load_async_to_lds_b128) &&           \
    __has_builtin(__builtin_amdgcn_s_wait_asynccnt)
#define HAVE_ASYNC_LDS 1
#else
#define HAVE_ASYNC_LDS 0
#endif

__device__ __forceinline__ void cp16_g2l(void* lds, const void* glb) {
#if HAVE_ASYNC_LDS
  // Builtin signature (per hipcc diagnostic): param 1 is AS1 (printed as
  // "__device__") pointer to int __vector(4); param 2 is the LDS (AS3) side.
  __builtin_amdgcn_global_load_async_to_lds_b128(
      (__attribute__((address_space(1))) v4i_t*)(void*)glb,
      (__attribute__((address_space(3))) v4i_t*)lds, 0, 0);
#else
  *(float4*)lds = *(const float4*)glb;
#endif
}

__device__ __forceinline__ void async_wait_all() {
#if HAVE_ASYNC_LDS
  __builtin_amdgcn_s_wait_asynccnt(0);
#endif
}

// ============================================================================
// WMMA GEMM:  C[M,N] = epilogue(A[M,K] @ W[K,N] + bias) (+ resid)
//   - row-major fp32 in/out; fp32 tiles in LDS; bf16 cvt in-register
//   - block tile 128x64x32, 8 waves, each wave = 32x32 (2x2 wmma frags)
//   - double-buffered async staging (tile k+1 DMA overlaps tile k WMMA)
//   - requires K%32==0, M%128==0, N%4==0 (true for every call here)
//   - rows padded +4 floats: keeps 16B chunks aligned for b128 async writes
//     and makes A-column / B-strided LDS reads bank-conflict-free
//   - ACT: 0 = identity, 1 = exact GELU (after bias, before resid)
// ============================================================================
#define BM 128
#define BN 64
#define BK 32

template <int ACT>
__global__ __launch_bounds__(256) void gemm_bias_kernel(
    const float* __restrict__ A, const float* __restrict__ W,
    const float* __restrict__ bias, const float* __restrict__ resid,
    float* __restrict__ C, int M, int N, int K) {
  __shared__ float sA[2][BM][BK + 4];   // 2*128*36*4 = 36.9KB
  __shared__ float sB[2][BK][BN + 4];   // 2*32*68*4  = 17.4KB

  const int tid  = threadIdx.x;
  const int lane = tid & 31;
  const int wave = tid >> 5;      // 0..7
  const int wm   = wave >> 1;     // 0..3  (M direction)
  const int wn   = wave & 1;      // 0..1  (N direction)
  const int bm   = blockIdx.y * BM;
  const int bn   = blockIdx.x * BN;

  const v8f vzero = {0.f, 0.f, 0.f, 0.f, 0.f, 0.f, 0.f, 0.f};
  v8f acc[2][2];
  acc[0][0] = vzero; acc[0][1] = vzero; acc[1][0] = vzero; acc[1][1] = vzero;

  // staging geometry: 16B chunks. A: 1024 chunks -> 4/thread; B: 512 -> 2.
  const int arow = tid >> 3;           // A row base (0..31), rows arow+32j
  const int acol = (tid & 7) * 4;      // A col (float) 0..28
  const int brow = tid >> 4;           // B row base (0..15), rows brow+16j
  const int bcol = (tid & 15) * 4;     // B col (float) 0..60

  // fragment geometry (ISA 7.12.2 layouts)
  const int am   = lane & 15;          // A frag: M within 16
  const int ak   = (lane >> 4) * 8;    // A frag: K group offset
  const int bn_l = lane & 15;          // B frag: N column
  const int bk   = (lane >> 4) * 8;    // B frag: K group offset

  auto stage = [&](int buf, int k0) {
#pragma unroll
    for (int j = 0; j < 4; ++j) {      // A tile: always fully in range
      const int r = arow + 32 * j;
      cp16_g2l(&sA[buf][r][acol], A + (size_t)(bm + r) * K + k0 + acol);
    }
    const int n0 = bn + bcol;
    const bool bok = (n0 + 4) <= N;    // chunk fully in or fully out (N%4==0)
#pragma unroll
    for (int j = 0; j < 2; ++j) {
      const int r = brow + 16 * j;
      if (bok) {
        cp16_g2l(&sB[buf][r][bcol], W + (size_t)(k0 + r) * N + n0);
      } else {
        const float4 z = {0.f, 0.f, 0.f, 0.f};
        *(float4*)&sB[buf][r][bcol] = z;
      }
    }
  };

  auto compute = [&](int buf) {
    v16bf av[2], bv[2];
#pragma unroll
    for (int i = 0; i < 2; ++i) {
      const float* pr = &sA[buf][wm * 32 + i * 16 + am][0];
      float t[16];
      *(float4*)&t[0]  = *(const float4*)(pr + ak);        // k = ak..ak+3
      *(float4*)&t[4]  = *(const float4*)(pr + ak + 4);    // k = ak+4..ak+7
      *(float4*)&t[8]  = *(const float4*)(pr + ak + 16);   // k = ak+16..
      *(float4*)&t[12] = *(const float4*)(pr + ak + 20);   // k = ak+20..
#pragma unroll
      for (int e = 0; e < 16; ++e) av[i][e] = (__bf16)t[e];
    }
#pragma unroll
    for (int j = 0; j < 2; ++j) {
      const int n = wn * 32 + j * 16 + bn_l;
#pragma unroll
      for (int e = 0; e < 16; ++e) bv[j][e] = (__bf16)sB[buf][bk + e + (e & 8)][n];
    }
#pragma unroll
    for (int i = 0; i < 2; ++i)
#pragma unroll
      for (int j = 0; j < 2; ++j)
        acc[i][j] = __builtin_amdgcn_wmma_f32_16x16x32_bf16(
            false, av[i], false, bv[j], (short)0, acc[i][j], false, false);
  };

  const int nk = K / BK;
  stage(0, 0);
  async_wait_all();
  __syncthreads();
  for (int kb = 0; kb < nk; ++kb) {
    const int cur = kb & 1;
    if (kb + 1 < nk) stage(cur ^ 1, (kb + 1) * BK);  // DMA next tile
    compute(cur);                                    // WMMA current tile
    async_wait_all();
    __syncthreads();
  }

  // ---- epilogue / store (C/D layout: vgpr r -> M=r or r+8, N=lane&15) ----
  const int mh = (lane >> 4) * 8;
#pragma unroll
  for (int i = 0; i < 2; ++i)
#pragma unroll
    for (int j = 0; j < 2; ++j) {
      const int n = bn + wn * 32 + j * 16 + (lane & 15);
      if (n >= N) continue;
#pragma unroll
      for (int r = 0; r < 8; ++r) {
        const int m = bm + wm * 32 + i * 16 + r + mh;
        float v = acc[i][j][r];
        if (bias) v += bias[n];
        if (ACT == 1) v = gelu_f(v);
        if (resid) v += resid[(size_t)m * N + n];
        C[(size_t)m * N + n] = v;
      }
    }
}

// ============================================================================
// Row LayerNorm over 768 cols.  One 256-thread block per row (3 elems/thread).
// Input row r maps through (rows_pb, bstride, rowoff) so the non-contiguous
// tail slice x[:,5:,:] can be read in place; x2 (optional, contiguous) is
// summed before normalization (for ln(a + h)).
// ============================================================================
__global__ __launch_bounds__(256) void ln_kernel(
    const float* __restrict__ x, const float* __restrict__ x2,
    const float* __restrict__ g, const float* __restrict__ beta,
    float* __restrict__ out, int rows_pb, int bstride, int rowoff) {
  __shared__ float red[256];
  const int r = blockIdx.x;
  const int b = r / rows_pb;
  const int i = r - b * rows_pb;
  const float* xr = x + ((size_t)b * bstride + rowoff + i) * DIM;
  const int tid = threadIdx.x;

  float v[3];
#pragma unroll
  for (int j = 0; j < 3; ++j) {
    int idx = tid + 256 * j;
    float t = xr[idx];
    if (x2) t += x2[(size_t)r * DIM + idx];
    v[j] = t;
  }
  red[tid] = v[0] + v[1] + v[2];
  __syncthreads();
  for (int o = 128; o > 0; o >>= 1) {
    if (tid < o) red[tid] += red[tid + o];
    __syncthreads();
  }
  const float mu = red[0] * (1.0f / 768.0f);
  __syncthreads();
  float d0 = v[0] - mu, d1 = v[1] - mu, d2 = v[2] - mu;
  red[tid] = d0 * d0 + d1 * d1 + d2 * d2;
  __syncthreads();
  for (int o = 128; o > 0; o >>= 1) {
    if (tid < o) red[tid] += red[tid + o];
    __syncthreads();
  }
  const float rstd = rsqrtf(red[0] * (1.0f / 768.0f) + 1e-6f);
#pragma unroll
  for (int j = 0; j < 3; ++j) {
    int idx = tid + 256 * j;
    out[(size_t)r * DIM + idx] = (v[j] - mu) * rstd * g[idx] + beta[idx];
  }
}

// ============================================================================
// Multi-scale deformable attention sampling (grid_sample align_corners=False,
// zero padding, mmcv semantics).  One wave per (b, q, head); lane owns
// channels {lane, lane+32, lane+64, lane+96} -> each corner gather is a
// contiguous 512B burst across the wave (L2-resident).
// ============================================================================
__global__ __launch_bounds__(256) void msda_kernel(
    const float* __restrict__ val,   // (B, Nv, 768) viewed as (B,Nv,6,128)
    const float* __restrict__ off,   // (B, Nq, 6*L*4*2)
    const float* __restrict__ awl,   // (B, Nq, 6*L*4)  softmax logits
    const float* __restrict__ ref,   // (B, Nq, 2)
    float* __restrict__ out,         // (B, Nq, 768)
    int Nq, int Nv, int L,
    int h0, int w0, int s0, int h1, int w1, int s1, int h2, int w2, int s2) {
  const int gw   = (blockIdx.x * blockDim.x + threadIdx.x) >> 5;
  const int lane = threadIdx.x & 31;
  const int total = BSZ * Nq * NHEAD;
  if (gw >= total) return;
  const int h = gw % NHEAD;
  const int q = (gw / NHEAD) % Nq;
  const int b = gw / (NHEAD * Nq);
  const int LP = L * 4;

  const float* offp = off + ((size_t)(b * Nq + q)) * (NHEAD * LP * 2) + h * (LP * 2);
  const float* awp  = awl + ((size_t)(b * Nq + q)) * (NHEAD * LP) + h * LP;

  // softmax over L*P (<=12) — computed redundantly per lane, trivial
  float lg[12];
  float mx = -3.4e38f;
  for (int i = 0; i < LP; ++i) { lg[i] = awp[i]; mx = fmaxf(mx, lg[i]); }
  float se = 0.f;
  for (int i = 0; i < LP; ++i) { lg[i] = __expf(lg[i] - mx); se += lg[i]; }
  const float inv = 1.0f / se;

  const float rx = ref[((size_t)(b * Nq + q)) * 2 + 0];
  const float ry = ref[((size_t)(b * Nq + q)) * 2 + 1];

  const int HH[3] = {h0, h1, h2};
  const int WW[3] = {w0, w1, w2};
  const int SS[3] = {s0, s1, s2};

  float a0 = 0.f, a1 = 0.f, a2 = 0.f, a3 = 0.f;
  for (int l = 0; l < L; ++l) {
    const int Hl = HH[l], Wl = WW[l], st = SS[l];
    const float fWl = (float)Wl, fHl = (float)Hl;
    for (int p = 0; p < 4; ++p) {
      const float ox = offp[(l * 4 + p) * 2 + 0];
      const float oy = offp[(l * 4 + p) * 2 + 1];
      const float fx = (rx + ox / fWl) * fWl - 0.5f;
      const float fy = (ry + oy / fHl) * fHl - 0.5f;
      const float x0 = floorf(fx), y0 = floorf(fy);
      const float wx = fx - x0, wy = fy - y0;
      const int ix0 = (int)x0, iy0 = (int)y0;
      const float aw = lg[l * 4 + p] * inv;
#pragma unroll
      for (int dy = 0; dy < 2; ++dy)
#pragma unroll
        for (int dx = 0; dx < 2; ++dx) {
          const int ix = ix0 + dx, iy = iy0 + dy;
          if (ix < 0 || ix >= Wl || iy < 0 || iy >= Hl) continue;  // zero pad
          const float wgt = (dy ? wy : 1.f - wy) * (dx ? wx : 1.f - wx) * aw;
          const float* vp =
              val + ((size_t)(b * Nv + st + iy * Wl + ix)) * DIM + h * HD + lane;
          a0 += wgt * vp[0];
          a1 += wgt * vp[32];
          a2 += wgt * vp[64];
          a3 += wgt * vp[96];
        }
    }
  }
  float* op = out + ((size_t)(b * Nq + q)) * DIM + h * HD + lane;
  op[0] = a0; op[32] = a1; op[64] = a2; op[96] = a3;
}

// ============================================================================
// Depthwise 3x3 ('SAME', zero pad) over the 3 pyramid segments, fused +bias
// and exact GELU.  Data stays in (B, N=5376, 192) token-major layout; the
// NCHW transpose in the reference is just an index remap here.
// ============================================================================
__global__ __launch_bounds__(256) void dwconv_gelu_kernel(
    const float* __restrict__ h, const float* __restrict__ w,
    const float* __restrict__ bias, float* __restrict__ out) {
  const int idx = blockIdx.x * blockDim.x + threadIdx.x;
  const int total = BSZ * NC * CCH;
  if (idx >= total) return;
  const int ch = idx % CCH;
  const int n  = (idx / CCH) % NC;
  const int b  = idx / (CCH * NC);
  int n0, hh, ww;
  if (n < 4096)      { n0 = 0;    hh = 64; ww = 64; }
  else if (n < 5120) { n0 = 4096; hh = 32; ww = 32; }
  else               { n0 = 5120; hh = 16; ww = 16; }
  const int r = n - n0;
  const int y = r / ww, x = r % ww;
  float acc = bias[ch];
#pragma unroll
  for (int dy = -1; dy <= 1; ++dy)
#pragma unroll
    for (int dx = -1; dx <= 1; ++dx) {
      const int yy = y + dy, xx = x + dx;
      if (yy < 0 || yy >= hh || xx < 0 || xx >= ww) continue;
      acc += w[ch * 9 + (dy + 1) * 3 + (dx + 1)] *
             h[((size_t)b * NC + n0 + yy * ww + xx) * CCH + ch];
    }
  out[idx] = gelu_f(acc);
}

// x_out[b,n] = x[b,n]  (n<5)  |  x[b,n] + gamma * y[b,n-5]  (n>=5)
__global__ __launch_bounds__(256) void assemble_x_kernel(
    const float* __restrict__ x, const float* __restrict__ y,
    const float* __restrict__ gamma, float* __restrict__ outx) {
  const int idx = blockIdx.x * blockDim.x + threadIdx.x;
  if (idx >= BSZ * NX * DIM) return;
  const int d = idx % DIM;
  const int n = (idx / DIM) % NX;
  const int b = idx / (DIM * NX);
  float v = x[idx];
  if (n >= 5) v += gamma[d] * y[((size_t)b * NT + (n - 5)) * DIM + d];
  outx[idx] = v;
}

__global__ __launch_bounds__(256) void add_kernel(
    const float* __restrict__ a, const float* __restrict__ b,
    float* __restrict__ o, int n) {
  const int idx = blockIdx.x * blockDim.x + threadIdx.x;
  if (idx < n) o[idx] = a[idx] + b[idx];
}

// ============================================================================
// Launcher.  Input order = setup_inputs() dict flattened (nested dicts in
// insertion order):
//  0 x  1 c  2 ref1  3 ref2
//  4 inj_qn_g 5 inj_qn_b 6 inj_fn_g 7 inj_fn_b 8 inj_gamma
//  inj_attn: 9 Wv 10 bv 11 Woff 12 boff 13 Waw 14 baw 15 Wout 16 bout
//            17 ln1_g 18 ln1_b 19 ln2_g 20 ln2_b 21 w1 22 b1 23 w2 24 b2
//  25 ext_qn_g 26 ext_qn_b 27 ext_fn_g 28 ext_fn_b
//  ext_attn: 29..44 (same 16-slot layout)
//  45 ext_ffnn_g 46 ext_ffnn_b
//  ext_cffn: 47 fc1_w 48 fc1_b 49 dw_w 50 dw_b 51 fc2_w 52 fc2_b
// ============================================================================
extern "C" void kernel_launch(void* const* d_in, const int* in_sizes, int n_in,
                              void* d_out, int out_size, void* d_ws,
                              size_t ws_size, hipStream_t stream) {
  (void)in_sizes; (void)n_in; (void)out_size; (void)ws_size;
#define PF(i) ((const float*)d_in[(i)])
  const float* x    = PF(0);
  const float* c    = PF(1);
  const float* ref1 = PF(2);
  const float* ref2 = PF(3);

  float* outx = (float*)d_out;                       // (4,1029,768)
  float* outc = outx + (size_t)BSZ * NX * DIM;       // (4,5376,768)

  // ---- workspace carving (~93M floats / ~354MiB) ----
  float* fnbuf  = (float*)d_ws;                         // 21504x768
  float* valbuf = fnbuf  + (size_t)BSZ * NC * DIM;      // 21504x768
  float* qnbuf  = valbuf + (size_t)BSZ * NC * DIM;      // 4096x768
  float* tbuf   = qnbuf  + (size_t)BSZ * NT * DIM;      // 21504x768
  float* abuf   = tbuf   + (size_t)BSZ * NC * DIM;      // 21504x768
  float* hbuf   = abuf   + (size_t)BSZ * NC * DIM;      // 21504x1024
  float* offbuf = hbuf   + (size_t)BSZ * NC * FFN;      // <=21504x48
  float* awbuf  = offbuf + (size_t)BSZ * NC * 48;       // <=21504x24

  const int MI = BSZ * NT;   // 4096 injector queries
  const int ME = BSZ * NC;   // 21504 extractor queries

#define GEMM(ACT, A_, W_, B_, R_, O_, M_, N_, K_)                              \
  gemm_bias_kernel<ACT><<<dim3(((N_) + BN - 1) / BN, (M_) / BM), 256, 0,       \
                          stream>>>((A_), (W_), (B_), (R_), (O_), (M_), (N_),  \
                                    (K_))

  // ======================= Injector =======================
  // qn = ln(x[:,5:]) ; fn = ln(c)
  ln_kernel<<<MI, 256, 0, stream>>>(x, nullptr, PF(4), PF(5), qnbuf, NT, NX, 5);
  ln_kernel<<<ME, 256, 0, stream>>>(c, nullptr, PF(6), PF(7), fnbuf, NC, NC, 0);
  // val = fn @ Wv + bv
  GEMM(0, fnbuf, PF(9), PF(10), nullptr, valbuf, ME, DIM, DIM);
  // off / aw logits
  GEMM(0, qnbuf, PF(11), PF(12), nullptr, offbuf, MI, 144, DIM);
  GEMM(0, qnbuf, PF(13), PF(14), nullptr, awbuf, MI, 72, DIM);
  // sampled = msda-core  (fn dead -> reuse fnbuf for the 4096x768 result)
  msda_kernel<<<(BSZ * NT * NHEAD) / 8, 256, 0, stream>>>(
      valbuf, offbuf, awbuf, ref1, fnbuf, NT, NC, 3,
      64, 64, 0, 32, 32, 4096, 16, 16, 5120);
  // t0 = sampled @ Wout + bout + qn  (fused residual)
  GEMM(0, fnbuf, PF(15), PF(16), qnbuf, tbuf, MI, DIM, DIM);
  // a = ln1(t0)
  ln_kernel<<<MI, 256, 0, stream>>>(tbuf, nullptr, PF(17), PF(18), abuf, MI, 0, 0);
  // h = gelu(a@w1+b1) @ w2 + b2
  GEMM(1, abuf, PF(21), PF(22), nullptr, hbuf, MI, FFN, DIM);
  GEMM(0, hbuf, PF(23), PF(24), nullptr, tbuf, MI, DIM, FFN);
  // y = ln2(a + h)
  ln_kernel<<<MI, 256, 0, stream>>>(abuf, tbuf, PF(19), PF(20), fnbuf, MI, 0, 0);
  // x_out = [x[:, :5], x[:,5:] + gamma*y]
  assemble_x_kernel<<<(BSZ * NX * DIM) / 256, 256, 0, stream>>>(x, fnbuf, PF(8),
                                                                outx);

  // ======================= Extractor =======================
  // qn2 = ln(c) ; fn2 = ln(x_out[:,5:])
  ln_kernel<<<ME, 256, 0, stream>>>(c, nullptr, PF(25), PF(26), fnbuf, NC, NC, 0);
  ln_kernel<<<MI, 256, 0, stream>>>(outx, nullptr, PF(27), PF(28), qnbuf, NT, NX, 5);
  // val2 = fn2 @ Wv
  GEMM(0, qnbuf, PF(29), PF(30), nullptr, valbuf, MI, DIM, DIM);
  // off2 / aw2
  GEMM(0, fnbuf, PF(31), PF(32), nullptr, offbuf, ME, 48, DIM);
  GEMM(0, fnbuf, PF(33), PF(34), nullptr, awbuf, ME, 24, DIM);
  // sampled2 -> tbuf
  msda_kernel<<<(BSZ * NC * NHEAD) / 8, 256, 0, stream>>>(
      valbuf, offbuf, awbuf, ref2, tbuf, NC, NT, 1,
      32, 32, 0, 0, 0, 0, 0, 0, 0);
  // t2 = sampled2 @ Wout + bout + qn2
  GEMM(0, tbuf, PF(35), PF(36), fnbuf, abuf, ME, DIM, DIM);
  // a2 = ln1(t2) -> tbuf
  ln_kernel<<<ME, 256, 0, stream>>>(abuf, nullptr, PF(37), PF(38), tbuf, ME, 0, 0);
  // h2 = gelu(a2@w1+b1)@w2+b2 -> fnbuf (qn2 dead)
  GEMM(1, tbuf, PF(41), PF(42), nullptr, hbuf, ME, FFN, DIM);
  GEMM(0, hbuf, PF(43), PF(44), nullptr, fnbuf, ME, DIM, FFN);
  // y2 = ln2(a2 + h2) -> abuf ; c1 = c + y2 -> out
  ln_kernel<<<ME, 256, 0, stream>>>(tbuf, fnbuf, PF(39), PF(40), abuf, ME, 0, 0);
  add_kernel<<<(ME * DIM) / 256, 256, 0, stream>>>(c, abuf, outc, ME * DIM);

  // ======================= Conv-FFN =======================
  // cn = ln(c1)
  ln_kernel<<<ME, 256, 0, stream>>>(outc, nullptr, PF(45), PF(46), tbuf, ME, 0, 0);
  // h = cn @ fc1 + b1
  GEMM(0, tbuf, PF(47), PF(48), nullptr, hbuf, ME, CCH, DIM);
  // hconv = gelu(dwconv3x3(h) + b)
  float* hconv = hbuf + (size_t)ME * CCH;
  dwconv_gelu_kernel<<<(ME * CCH) / 256, 256, 0, stream>>>(hbuf, PF(49), PF(50),
                                                           hconv);
  // c_out = c1 + hconv @ fc2 + b2   (residual read+write on outc is per-element)
  GEMM(0, hconv, PF(51), PF(52), outc, outc, ME, DIM, CCH);

#undef GEMM
#undef PF
}
